// CausalMultiHeadAttention_90984587199153
// MI455X (gfx1250) — matrix operand order repaired and checked by reading, not compiled, not run on hardware
//
#include <hip/hip_runtime.h>

typedef __bf16 bf16_t;
typedef __attribute__((ext_vector_type(16))) __bf16 v16bf;
typedef __attribute__((ext_vector_type(8)))  __bf16 v8bf;
typedef __attribute__((ext_vector_type(8)))  float  v8f;
typedef unsigned int u32x4 __attribute__((ext_vector_type(4)));
typedef int          i32x4 __attribute__((ext_vector_type(4)));
typedef int          i32x8 __attribute__((ext_vector_type(8)));

#define SEQ    2048
#define NBATCH 2
#define NHEADS 16
#define DHEAD  128
#define DMODEL 2048
#define EQKV   (3 * DMODEL)
#define MTOK   (SEQ * NBATCH)   // 4096 tokens

// LDS panel row stride: 128 data elems + 8 pad elems (16B) -> 272B/row.
// 272B = 68 DWORDs -> successive rows shift 4 banks; lanes 0-15 of a
// fragment load hit 16 distinct bank groups (no conflicts).
#define LDP 136
// Flash P tile stride: 32 data + 8 pad = 40 elems (80B = 20 DWORD shift).
#define PLD 40

#if defined(__has_builtin)
#if __has_builtin(__builtin_amdgcn_tensor_load_to_lds) && \
    __has_builtin(__builtin_amdgcn_s_wait_tensorcnt)
#define HAVE_TDM 1
#endif
#endif
#ifndef HAVE_TDM
#define HAVE_TDM 0
#endif

// ---------- helpers ----------

__device__ __forceinline__ bf16_t f2bf(float f) {
  unsigned u = __builtin_bit_cast(unsigned, f);
  unsigned r = u + 0x7FFFu + ((u >> 16) & 1u);   // round-to-nearest-even
  unsigned short h = (unsigned short)(r >> 16);
  return __builtin_bit_cast(bf16_t, h);
}

__device__ __forceinline__ v8f v8f_zero() {
  v8f z;
#pragma unroll
  for (int i = 0; i < 8; ++i) z[i] = 0.0f;
  return z;
}

__device__ __forceinline__ v16bf combine16(v8bf lo, v8bf hi) {
  v16bf r;
#pragma unroll
  for (int i = 0; i < 8; ++i) { r[i] = lo[i]; r[i + 8] = hi[i]; }
  return r;
}

// A fragment (16x32 bf16) from row-major source, leading dim ld (ISA 7.12.2):
// lanes 0-15 row m=lane, K in {0..7,16..23}; lanes 16-31 same rows, K {8..15,24..31}.
__device__ __forceinline__ v16bf load_a_frag(const bf16_t* base, int ld, int lane) {
  int m = lane & 15, half = lane >> 4;
  const bf16_t* p = base + (size_t)m * ld + half * 8;
  v8bf lo = *(const v8bf*)(p);
  v8bf hi = *(const v8bf*)(p + 16);
  return combine16(lo, hi);
}

// B fragment (32x16 bf16) from the *transposed* operand Bt (N x K row-major):
// lane holds column n, K in {0..7,16..23} (lanes 0-15) or {8..15,24..31} (lanes 16-31).
__device__ __forceinline__ v16bf load_b_frag(const bf16_t* base, int ld, int lane) {
  int n = lane & 15, half = lane >> 4;
  const bf16_t* p = base + (size_t)n * ld + half * 8;
  v8bf lo = *(const v8bf*)(p);
  v8bf hi = *(const v8bf*)(p + 16);
  return combine16(lo, hi);
}

__device__ __forceinline__ v8f wmma_bf(v16bf a, v16bf b, v8f c) {
  return __builtin_amdgcn_wmma_f32_16x16x32_bf16(false, a, false, b,
                                                 (short)0, c, false, false);
}

// ---------- Tensor Data Mover: 2D bf16 tile (global -> LDS) ----------
// D# per CDNA5 ISA. With pad=true, TDM inserts 16B of
// LDS padding after every 256B row (pad_interval=5 -> 64 DWORDs,
// pad_amount=3 -> 4 DWORDs), producing the bank-conflict-free LDP layout.
#if HAVE_TDM
__device__ __forceinline__ void tdm_load_2d(unsigned lds_off, const void* gptr,
                                            unsigned tile_d0, unsigned tile_d1,
                                            unsigned tensor_d0, unsigned tensor_d1,
                                            unsigned stride0, bool pad) {
  unsigned long long ga = (unsigned long long)(uintptr_t)gptr;
  u32x4 g0;
  g0[0] = 1u;                                        // count=1, user descriptor
  g0[1] = lds_off;                                   // LDS byte address
  g0[2] = (unsigned)(ga & 0xFFFFFFFFu);              // global_addr[31:0]
  g0[3] = (unsigned)((ga >> 32) & 0x01FFFFFFu) | (2u << 30);  // [56:32] | type=2
  i32x8 g1;
  unsigned w0 = (1u << 16);                          // data_size=1 (2 bytes)
  if (pad) w0 |= (1u << 20) | (5u << 22) | (3u << 25);
  g1[0] = (int)w0;
  g1[1] = (int)((tensor_d0 & 0xFFFFu) << 16);        // tensor_dim0[15:0]
  g1[2] = (int)(((tensor_d0 >> 16) & 0xFFFFu) | ((tensor_d1 & 0xFFFFu) << 16));
  g1[3] = (int)(((tensor_d1 >> 16) & 0xFFFFu) | ((tile_d0 & 0xFFFFu) << 16));
  g1[4] = (int)(tile_d1 & 0xFFFFu);                  // tile_dim1 (tile_dim2=0)
  g1[5] = (int)stride0;                              // tensor_dim0_stride[31:0]
  g1[6] = 0;
  g1[7] = 0;
  i32x4 z4 = {0, 0, 0, 0};
#if __clang_major__ >= 23
  i32x8 z8 = {0, 0, 0, 0, 0, 0, 0, 0};
  __builtin_amdgcn_tensor_load_to_lds(g0, g1, z4, z4, z8, 0);
#else
  __builtin_amdgcn_tensor_load_to_lds(g0, g1, z4, z4, 0);
#endif
}
#endif

// Fallback cooperative panel copy: 64 rows x 128 bf16 -> padded LDP layout.
__device__ __forceinline__ void stage_panel(bf16_t* dst, const bf16_t* src, int ld) {
  for (int i = threadIdx.x; i < 64 * 16; i += 128) {   // 16 x 16B chunks per row
    int r = i >> 4, c = (i & 15) * 8;
    *(v8bf*)(dst + r * LDP + c) = *(const v8bf*)(src + (size_t)r * ld + c);
  }
}

// ---------- kernel 0: fp32 -> bf16 ----------

__global__ void cvt_f32_bf16(const float* __restrict__ src,
                             bf16_t* __restrict__ dst, int n) {
  for (int i = blockIdx.x * blockDim.x + threadIdx.x; i < n;
       i += gridDim.x * blockDim.x)
    dst[i] = f2bf(src[i]);
}

// ---------- shared GEMM core: 64x64 block tile, TDM double-buffered ----------
// Block = 4 waves; wave w owns rows m0+16w, all 64 columns. K panels of 128.

#if HAVE_TDM
#define TDM_ISSUE(KP, BUF)                                                       \
  if (wave == 0) {                                                               \
    tdm_load_2d((unsigned)(uintptr_t)&sA[(BUF)][0],                              \
                aBase + (size_t)(KP) * 128, 128, 64, (unsigned)DMODEL,           \
                (unsigned)aRows, (unsigned)DMODEL, true);                        \
    tdm_load_2d((unsigned)(uintptr_t)&sB[(BUF)][0],                              \
                bBase + (size_t)(KP) * 128, 128, 64, (unsigned)DMODEL,           \
                (unsigned)bRows, (unsigned)DMODEL, true);                        \
  }
#define TDM_WAIT(MORE)                                                           \
  do {                                                                           \
    if (MORE) __builtin_amdgcn_s_wait_tensorcnt(2);                              \
    else      __builtin_amdgcn_s_wait_tensorcnt(0);                              \
  } while (0)
#else
#define TDM_ISSUE(KP, BUF)
#define TDM_WAIT(MORE)
#endif

#define GEMM_CORE(ACC, NPANEL)                                                   \
  __shared__ __align__(16) bf16_t sA[2][64 * LDP];                               \
  __shared__ __align__(16) bf16_t sB[2][64 * LDP];                               \
  v8f ACC[4];                                                                    \
  _Pragma("unroll") for (int j = 0; j < 4; ++j) ACC[j] = v8f_zero();             \
  {                                                                              \
    if (HAVE_TDM) {                                                              \
      TDM_ISSUE(0, 0);                                                           \
    }                                                                            \
    for (int kp = 0; kp < (NPANEL); ++kp) {                                      \
      if (HAVE_TDM) {                                                            \
        if (wave == 0) {                                                         \
          if (kp + 1 < (NPANEL)) { TDM_ISSUE(kp + 1, (kp + 1) & 1); }            \
          TDM_WAIT(kp + 1 < (NPANEL));                                           \
        }                                                                        \
        __syncthreads();                                                         \
      } else {                                                                   \
        __syncthreads();                                                         \
        stage_panel(sA[kp & 1], aBase + (size_t)kp * 128, DMODEL);               \
        stage_panel(sB[kp & 1], bBase + (size_t)kp * 128, DMODEL);               \
        __syncthreads();                                                         \
      }                                                                          \
      const bf16_t* pa = sA[kp & 1] + wave * 16 * LDP;                           \
      const bf16_t* pb = sB[kp & 1];                                             \
      _Pragma("unroll") for (int ks = 0; ks < 4; ++ks) {                         \
        v16bf a  = load_a_frag(pa + ks * 32, LDP, lane);                         \
        v16bf b0 = load_b_frag(pb + 0 * 16 * LDP + ks * 32, LDP, lane);          \
        v16bf b1 = load_b_frag(pb + 1 * 16 * LDP + ks * 32, LDP, lane);          \
        v16bf b2 = load_b_frag(pb + 2 * 16 * LDP + ks * 32, LDP, lane);          \
        v16bf b3 = load_b_frag(pb + 3 * 16 * LDP + ks * 32, LDP, lane);          \
        ACC[0] = wmma_bf(a, b0, ACC[0]);                                         \
        ACC[1] = wmma_bf(a, b1, ACC[1]);                                         \
        ACC[2] = wmma_bf(a, b2, ACC[2]);                                         \
        ACC[3] = wmma_bf(a, b3, ACC[3]);                                         \
      }                                                                          \
      __syncthreads();                                                           \
    }                                                                            \
  }

// ---------- kernel 1: QKV GEMM (NT) + bias + RoPE + scatter ----------

__global__ __launch_bounds__(128) void gemm_qkv_rope(
    const bf16_t* __restrict__ xb, const bf16_t* __restrict__ wb,
    const float* __restrict__ bias, const float* __restrict__ freqs,
    bf16_t* __restrict__ Qb, bf16_t* __restrict__ Kb, bf16_t* __restrict__ Vtb) {
  int lane = threadIdx.x & 31;
  int wave = threadIdx.x >> 5;
  int e0 = blockIdx.x * 64;           // 96 column blocks
  int m0 = blockIdx.y * 64;           // 64 row blocks
  const bf16_t* aBase = xb + (size_t)m0 * DMODEL;
  const bf16_t* bBase = wb + (size_t)e0 * DMODEL;
  const int aRows = MTOK, bRows = EQKV;
  (void)aRows; (void)bRows;

  GEMM_CORE(acc, DMODEL / 128)

  int mw = m0 + wave * 16;
  int n = lane & 15, half = lane >> 4;
#pragma unroll
  for (int j = 0; j < 4; ++j) {
    int e = e0 + j * 16 + n;
    float bv = bias[e];
    int section = e >> 11;            // 0=q, 1=k, 2=v (uniform per subtile)
    int c = e & (DMODEL - 1);
    int hh = c >> 7, d = c & (DHEAD - 1);
#pragma unroll
    for (int g = 0; g < 8; ++g) {
      int m = mw + g + 8 * half;      // token index (x is (S,B,D): m = s*B + b)
      int s = m >> 1, bat = m & 1;
      float val = acc[j][g] + bv;
      if (section < 2) {
        float f = freqs[s * (DHEAD / 2) + (d >> 1)];
        float cs = __cosf(f), sn = __sinf(f);
        float partner = __shfl_xor(val, 1, 32);   // pair lanes n, n^1
        val = (d & 1) ? (partner * sn + val * cs)   // odd:  a*sin + b*cos
                      : (val * cs - partner * sn);  // even: a*cos - b*sin
        bf16_t* dst = (section == 0) ? Qb : Kb;
        dst[(((size_t)(bat * NHEADS + hh)) * SEQ + s) * DHEAD + d] = f2bf(val);
      } else {
        Vtb[(((size_t)(bat * NHEADS + hh)) * DHEAD + d) * SEQ + s] = f2bf(val);
      }
    }
  }
}

// ---------- kernel 2: flash attention, one 16-row q-tile per wave ----------

__global__ __launch_bounds__(128) void flash_attn(
    const bf16_t* __restrict__ Qb, const bf16_t* __restrict__ Kb,
    const bf16_t* __restrict__ Vtb, bf16_t* __restrict__ Ob) {
  int lane = threadIdx.x & 31;
  int wave = threadIdx.x >> 5;
  int bh = blockIdx.y;
  int b = bh >> 4, h = bh & 15;
  int q0 = (blockIdx.x * 4 + wave) * 16;

  const bf16_t* Qh = Qb + (size_t)bh * SEQ * DHEAD;
  const bf16_t* Kh = Kb + (size_t)bh * SEQ * DHEAD;
  const bf16_t* Vh = Vtb + (size_t)bh * DHEAD * SEQ;

  __shared__ __align__(16) bf16_t plds[4][16 * PLD];  // per-wave P staging
  bf16_t* pw = plds[wave];

  v16bf qa[4];
#pragma unroll
  for (int kc = 0; kc < 4; ++kc)
    qa[kc] = load_a_frag(Qh + (size_t)q0 * DHEAD + kc * 32, DHEAD, lane);

  v8f o[8];
#pragma unroll
  for (int j = 0; j < 8; ++j) o[j] = v8f_zero();
  float rmax[8], rsum[8];
#pragma unroll
  for (int g = 0; g < 8; ++g) { rmax[g] = -3.0e38f; rsum[g] = 0.0f; }

  const float scale = 0.08838834764831845f;   // 1/sqrt(128)
  int n = lane & 15, half = lane >> 4;
  int ktiles = (q0 + 15) / 32 + 1;            // causal bound

  for (int kt = 0; kt < ktiles; ++kt) {
    int k0 = kt * 32;
    v8f s0 = v8f_zero(), s1 = v8f_zero();
#pragma unroll
    for (int kc = 0; kc < 4; ++kc) {
      v16bf kb0 = load_b_frag(Kh + (size_t)k0 * DHEAD + kc * 32, DHEAD, lane);
      v16bf kb1 = load_b_frag(Kh + (size_t)(k0 + 16) * DHEAD + kc * 32, DHEAD, lane);
      s0 = wmma_bf(qa[kc], kb0, s0);
      s1 = wmma_bf(qa[kc], kb1, s1);
    }

    float a0[8], a1[8], nm[8], alpha[8];
#pragma unroll
    for (int g = 0; g < 8; ++g) {
      int q = q0 + g + 8 * half;
      float v0 = s0[g] * scale;
      float v1 = s1[g] * scale;
      if (k0 + n > q)      v0 = -3.0e38f;     // causal mask
      if (k0 + 16 + n > q) v1 = -3.0e38f;
      a0[g] = v0; a1[g] = v1;
    }
#pragma unroll
    for (int g = 0; g < 8; ++g) {
      float mx = fmaxf(a0[g], a1[g]);
      mx = fmaxf(mx, __shfl_xor(mx, 1, 32));
      mx = fmaxf(mx, __shfl_xor(mx, 2, 32));
      mx = fmaxf(mx, __shfl_xor(mx, 4, 32));
      mx = fmaxf(mx, __shfl_xor(mx, 8, 32));
      nm[g] = fmaxf(rmax[g], mx);
    }
#pragma unroll
    for (int g = 0; g < 8; ++g) {
      alpha[g] = __expf(rmax[g] - nm[g]);
      float p0 = __expf(a0[g] - nm[g]);
      float p1 = __expf(a1[g] - nm[g]);
      float rs = p0 + p1;
      rs += __shfl_xor(rs, 1, 32);
      rs += __shfl_xor(rs, 2, 32);
      rs += __shfl_xor(rs, 4, 32);
      rs += __shfl_xor(rs, 8, 32);
      rsum[g] = rsum[g] * alpha[g] + rs;
      rmax[g] = nm[g];
      a0[g] = p0; a1[g] = p1;
    }
    // C-layout -> A-layout via per-wave LDS round trip (in-order within wave)
#pragma unroll
    for (int g = 0; g < 8; ++g) {
      int m = g + 8 * half;
      pw[m * PLD + n]      = f2bf(a0[g]);
      pw[m * PLD + 16 + n] = f2bf(a1[g]);
    }
    v16bf pa = load_a_frag(pw, PLD, lane);
#pragma unroll
    for (int j = 0; j < 8; ++j) {
      v8f oj = o[j];
#pragma unroll
      for (int g = 0; g < 8; ++g) oj[g] *= alpha[g];
      v16bf vb = load_b_frag(Vh + (size_t)(j * 16) * SEQ + k0, SEQ, lane);
      o[j] = wmma_bf(pa, vb, oj);
    }
  }

#pragma unroll
  for (int j = 0; j < 8; ++j) {
#pragma unroll
    for (int g = 0; g < 8; ++g) {
      int m = g + 8 * half;
      int srow = q0 + m;
      float val = o[j][g] / rsum[g];
      int d = j * 16 + n;
      Ob[((size_t)(srow * NBATCH + b)) * DMODEL + h * DHEAD + d] = f2bf(val);
    }
  }
}

// ---------- kernel 3: output GEMM (NT) + bias -> fp32 d_out ----------

__global__ __launch_bounds__(128) void gemm_out(
    const bf16_t* __restrict__ Ob, const bf16_t* __restrict__ wob,
    const float* __restrict__ bias, float* __restrict__ out) {
  int lane = threadIdx.x & 31;
  int wave = threadIdx.x >> 5;
  int e0 = blockIdx.x * 64;           // 32 column blocks
  int m0 = blockIdx.y * 64;           // 64 row blocks
  const bf16_t* aBase = Ob + (size_t)m0 * DMODEL;
  const bf16_t* bBase = wob + (size_t)e0 * DMODEL;
  const int aRows = MTOK, bRows = DMODEL;
  (void)aRows; (void)bRows;

  GEMM_CORE(acc, DMODEL / 128)

  int mw = m0 + wave * 16;
  int n = lane & 15, half = lane >> 4;
#pragma unroll
  for (int j = 0; j < 4; ++j) {
    int e = e0 + j * 16 + n;
    float bv = bias[e];
#pragma unroll
    for (int g = 0; g < 8; ++g) {
      int m = mw + g + 8 * half;
      out[(size_t)m * DMODEL + e] = acc[j][g] + bv;   // (S,B,D) flat == token-major
    }
  }
}

// ---------- launcher ----------

extern "C" void kernel_launch(void* const* d_in, const int* in_sizes, int n_in,
                              void* d_out, int out_size, void* d_ws, size_t ws_size,
                              hipStream_t stream) {
  (void)in_sizes; (void)n_in; (void)out_size; (void)ws_size;
  const float* x     = (const float*)d_in[0];   // (S, B, D)
  const float* freqs = (const float*)d_in[1];   // (S, 64)
  const float* Wqkv  = (const float*)d_in[2];   // (6144, 2048)
  const float* bqkv  = (const float*)d_in[3];
  const float* Wo    = (const float*)d_in[4];   // (2048, 2048)
  const float* bo    = (const float*)d_in[5];
  float* out = (float*)d_out;

  char* ws = (char*)d_ws;
  size_t off = 0;
  bf16_t* xb    = (bf16_t*)(ws + off); off += (size_t)MTOK * DMODEL * 2;
  bf16_t* wqkvb = (bf16_t*)(ws + off); off += (size_t)EQKV * DMODEL * 2;
  bf16_t* wob   = (bf16_t*)(ws + off); off += (size_t)DMODEL * DMODEL * 2;
  bf16_t* Qb    = (bf16_t*)(ws + off); off += (size_t)NBATCH * NHEADS * SEQ * DHEAD * 2;
  bf16_t* Kb    = (bf16_t*)(ws + off); off += (size_t)NBATCH * NHEADS * SEQ * DHEAD * 2;
  bf16_t* Vtb   = (bf16_t*)(ws + off); off += (size_t)NBATCH * NHEADS * DHEAD * SEQ * 2;
  bf16_t* Ob    = (bf16_t*)(ws + off); off += (size_t)MTOK * DMODEL * 2;

  cvt_f32_bf16<<<4096, 256, 0, stream>>>(x, xb, MTOK * DMODEL);
  cvt_f32_bf16<<<4096, 256, 0, stream>>>(Wqkv, wqkvb, EQKV * DMODEL);
  cvt_f32_bf16<<<4096, 256, 0, stream>>>(Wo, wob, DMODEL * DMODEL);

  gemm_qkv_rope<<<dim3(EQKV / 64, MTOK / 64), 128, 0, stream>>>(
      xb, wqkvb, bqkv, freqs, Qb, Kb, Vtb);

  flash_attn<<<dim3(SEQ / 16 / 4, NBATCH * NHEADS), 128, 0, stream>>>(
      Qb, Kb, Vtb, Ob);

  gemm_out<<<dim3(DMODEL / 64, MTOK / 64), 128, 0, stream>>>(
      Ob, wob, bo, out);
}
